// MiniLSTM_1546188226597
// MI455X (gfx1250) — matrix operand order repaired and checked by reading, not compiled, not run on hardware
//
#include <hip/hip_runtime.h>
#include <hip/hip_bf16.h>

// ---------------------------------------------------------------------------
// MiniLSTM forward for MI455X (gfx1250, wave32, WMMA).
// V=2048 E=128 H=256 B=64 T=1024.
// Strategy:
//  * All GEMMs in bf16 WMMA (v_wmma_f32_16x16x32_bf16), fp32 accumulate.
//  * Sequential LSTM recurrence as a persistent 16-WG kernel with W_hh^T in
//    LDS, fp32 cell state in registers, device-scope atomic barrier per step.
//  * Logits GEMM streams 512MB of fp32 with nontemporal stores (HBM-bound).
// ---------------------------------------------------------------------------

typedef __attribute__((ext_vector_type(16))) __bf16    v16bf;
typedef __attribute__((ext_vector_type(8)))  float     v8f;
typedef __attribute__((ext_vector_type(4)))  unsigned  v4u;

static constexpr int Vv  = 2048;
static constexpr int Ee  = 128;
static constexpr int Hh  = 256;
static constexpr int Bb  = 64;
static constexpr int Tt  = 1024;
static constexpr int G4  = 4 * Hh;          // 1024
static constexpr int Mrows = Bb * Tt;       // 65536
static constexpr int NWG_LSTM = 16;

// ---------------- bf16 helpers (round-to-nearest-even, bit level) ----------
__device__ __forceinline__ __bf16 f2bf(float f) {
  unsigned u = __builtin_bit_cast(unsigned, f);
  unsigned r = (u + 0x7fffu + ((u >> 16) & 1u)) >> 16;
  return __builtin_bit_cast(__bf16, (unsigned short)r);
}
__device__ __forceinline__ float bf2f(__bf16 h) {
  unsigned u = ((unsigned)__builtin_bit_cast(unsigned short, h)) << 16;
  return __builtin_bit_cast(float, u);
}
__device__ __forceinline__ float sigmf(float x) {
  return 1.0f / (1.0f + __expf(-x));
}

// Build a v16bf from two 16-byte chunks.
struct TwoQ { v4u a, b; };
__device__ __forceinline__ v16bf load2x16B(const __bf16* p0, const __bf16* p1) {
  TwoQ s;
  s.a = *(const v4u*)p0;
  s.b = *(const v4u*)p1;
  return __builtin_bit_cast(v16bf, s);
}

// A-operand tile (16x32 bf16) from row-major A[*, lda], rows r0..r0+15, k0..k0+31.
// Lane L<16: M=L, K chunks {k0..k0+7, k0+16..k0+23}; lane>=16: +8 on K chunks.
__device__ __forceinline__ v16bf load_a_tile(const __bf16* A, int lda, int r0,
                                             int k0, int lane) {
  int row = r0 + (lane & 15);
  int ko  = k0 + ((lane >> 4) << 3);
  const __bf16* p = A + (size_t)row * lda + ko;
  return load2x16B(p, p + 16);
}

// B-operand tile (32x16 bf16) from K-major Bt[K][ldb], N=c0+lane%16, K chunks as A.
__device__ __forceinline__ v16bf load_b_tile(const __bf16* Bt, int ldb, int c0,
                                             int k0, int lane) {
  const __bf16* p = Bt + (size_t)(k0 + ((lane >> 4) << 3)) * ldb + c0 + (lane & 15);
  v16bf b;
  #pragma unroll
  for (int s = 0; s < 8; ++s) {
    b[s] = p[(size_t)s * ldb];
    b[s + 8] = p[(size_t)(s + 16) * ldb];
  }
  return b;
}

__device__ __forceinline__ v8f wmma_bf16(v16bf a, v16bf b, v8f c) {
  return __builtin_amdgcn_wmma_f32_16x16x32_bf16(false, a, false, b,
                                                 (short)0, c, false, false);
}

// ---------------- small utility kernels ------------------------------------
__global__ void zero_state_kernel(__bf16* hcur, unsigned* cnt, int ncnt) {
  int i = blockIdx.x * blockDim.x + threadIdx.x;
  if (i < 2 * Bb * Hh) hcur[i] = __builtin_bit_cast(__bf16, (unsigned short)0);
  if (i < ncnt) cnt[i] = 0u;
}

// W [N][K] fp32 row-major  ->  Wt [K][N] bf16 (K-major, matches WMMA B layout)
__global__ void transpose_bf16_kernel(const float* __restrict__ W,
                                      __bf16* __restrict__ Wt, int N, int K) {
  int i = blockIdx.x * blockDim.x + threadIdx.x;
  if (i >= N * K) return;
  int n = i / K, k = i - n * K;
  Wt[(size_t)k * N + n] = f2bf(W[i]);
}

// Embedding gather: xemb[m][0..127] = bf16(embed[x[m]][:]), 4 elems per thread.
__global__ void embed_kernel(const int* __restrict__ x,
                             const float* __restrict__ embed,
                             __bf16* __restrict__ xe) {
  int i = blockIdx.x * blockDim.x + threadIdx.x;
  int m = i >> 5;
  int e = (i & 31) * 4;
  if (m >= Mrows) return;
  int idx = x[m];
  float4 f = *(const float4*)(embed + (size_t)idx * Ee + e);
  __bf16* p = xe + (size_t)m * Ee + e;
  p[0] = f2bf(f.x); p[1] = f2bf(f.y); p[2] = f2bf(f.z); p[3] = f2bf(f.w);
}

// ---------------- generic WMMA GEMM: C[M,N] = A[M,K] * Bt[K,N] + bias -------
// WG = 256 threads = 8 waves, WG tile 64x128, wave tile 32x32 (4 WMMA tiles).
template <bool OUT_BF16>
__global__ void gemm_wmma_kernel(const __bf16* __restrict__ A,
                                 const __bf16* __restrict__ Bt,
                                 const float* __restrict__ bias1,
                                 const float* __restrict__ bias2,
                                 void* __restrict__ Cout,
                                 int M, int N, int K) {
  int lane = threadIdx.x & 31;
  int w = threadIdx.x >> 5;
  int r0 = blockIdx.y * 64 + (w >> 2) * 32;
  int c0 = blockIdx.x * 128 + (w & 3) * 32;

  v8f acc00 = {}, acc01 = {}, acc10 = {}, acc11 = {};
  for (int k0 = 0; k0 < K; k0 += 32) {
    v16bf a0 = load_a_tile(A, K, r0, k0, lane);
    v16bf a1 = load_a_tile(A, K, r0 + 16, k0, lane);
    v16bf b0 = load_b_tile(Bt, N, c0, k0, lane);
    v16bf b1 = load_b_tile(Bt, N, c0 + 16, k0, lane);
    acc00 = wmma_bf16(a0, b0, acc00);
    acc01 = wmma_bf16(a0, b1, acc01);
    acc10 = wmma_bf16(a1, b0, acc10);
    acc11 = wmma_bf16(a1, b1, acc11);
  }

  // Epilogue: element (r,lane) of each tile -> m = r0t + (lane>=16?8:0)+r,
  // n = c0t + (lane&15).
  v8f accs[4] = {acc00, acc01, acc10, acc11};
  #pragma unroll
  for (int t = 0; t < 4; ++t) {
    int rt = r0 + (t >> 1) * 16;
    int ct = c0 + (t & 1) * 16;
    int n  = ct + (lane & 15);
    float bias = (bias1 ? bias1[n] : 0.0f) + (bias2 ? bias2[n] : 0.0f);
    int mb = rt + ((lane >> 4) << 3);
    #pragma unroll
    for (int r = 0; r < 8; ++r) {
      float v = accs[t][r] + bias;
      size_t off = (size_t)(mb + r) * N + n;
      if (OUT_BF16) {
        ((__bf16*)Cout)[off] = f2bf(v);
      } else {
        __builtin_nontemporal_store(v, (float*)Cout + off);  // 512MB stream
      }
    }
  }
}

// ---------------- persistent LSTM layer kernel ------------------------------
// Grid: 16 WGs x 256 threads. WG wg owns H-columns [wg*16, wg*16+16) for all
// four gates. W_hh^T slice (256 K x 64 gate-cols) lives in LDS. fp32 cell
// state in registers (4 per thread). h exchanged through L2 (double-buffered
// hcur), device-scope release/acquire atomic barrier per timestep.
__global__ void lstm_layer_kernel(const __bf16* __restrict__ G,    // [B*T][4H]
                                  const __bf16* __restrict__ Whht, // [H][4H]
                                  __bf16* __restrict__ Hseq,       // [B*T][H]
                                  __bf16* __restrict__ hcur,       // [2][B][H]
                                  unsigned* __restrict__ counter,
                                  float* __restrict__ hT,          // [B][H]
                                  float* __restrict__ cT) {        // [B][H]
  __shared__ __align__(32) __bf16 lds_w[4][Hh][16];   // 32 KB
  __shared__ float lds_g[4][Bb][16];                  // 16 KB

  const int wg   = blockIdx.x;            // 0..15
  const int c0   = wg * 16;               // H-column base
  const int tid  = threadIdx.x;
  const int lane = tid & 31;
  const int w    = tid >> 5;
  const int gate_w = w >> 1;              // this wave's gate (0..3)
  const int mbase  = (w & 1) * 32;        // batch rows 0..31 or 32..63

  // Stage W_hh^T slice into LDS: lds_w[gate][k][n] = Whht[k][gate*H + c0 + n].
  for (int e = tid; e < 4 * Hh * 16; e += 256) {
    int gate = e >> 12;
    int k = (e >> 4) & 255;
    int n = e & 15;
    lds_w[gate][k][n] = Whht[(size_t)k * G4 + gate * Hh + c0 + n];
  }
  __syncthreads();

  // Update-phase ownership: thread -> batch row urow, 4 consecutive cols.
  const int urow = tid >> 2;
  const int ucol = (tid & 3) * 4;
  float creg[4] = {0.f, 0.f, 0.f, 0.f};

  for (int t = 0; t < Tt; ++t) {
    const __bf16* hprev = hcur + (size_t)(t & 1) * (Bb * Hh);
    __bf16* hnext = hcur + (size_t)((t + 1) & 1) * (Bb * Hh);

    // ---- recurrent GEMM: [64,256] x [256, 64 gate cols] for this WG -------
    v8f acc0 = {}, acc1 = {};
    #pragma unroll
    for (int k0 = 0; k0 < Hh; k0 += 32) {
      const int kb = k0 + ((lane >> 4) << 3);
      v16bf bt;                                            // from LDS
      #pragma unroll
      for (int s = 0; s < 8; ++s) {
        bt[s] = lds_w[gate_w][kb + s][lane & 15];
        bt[s + 8] = lds_w[gate_w][kb + 16 + s][lane & 15];
      }
      v16bf a0 = load_a_tile(hprev, Hh, mbase, k0, lane);  // from L2
      v16bf a1 = load_a_tile(hprev, Hh, mbase + 16, k0, lane);
      acc0 = wmma_bf16(a0, bt, acc0);
      acc1 = wmma_bf16(a1, bt, acc1);
    }
    {
      int n  = lane & 15;
      int mb = (lane >> 4) << 3;
      #pragma unroll
      for (int r = 0; r < 8; ++r) {
        lds_g[gate_w][mbase + mb + r][n]      = acc0[r];
        lds_g[gate_w][mbase + 16 + mb + r][n] = acc1[r];
      }
    }
    __syncthreads();

    // ---- gate nonlinearities + state update -------------------------------
    const __bf16* gp = G + ((size_t)urow * Tt + t) * G4;
    __bf16* hs = Hseq + ((size_t)urow * Tt + t) * Hh;
    #pragma unroll
    for (int jj = 0; jj < 4; ++jj) {
      int j = ucol + jj;
      float ipre = lds_g[0][urow][j] + bf2f(gp[0 * Hh + c0 + j]);
      float fpre = lds_g[1][urow][j] + bf2f(gp[1 * Hh + c0 + j]);
      float gpre = lds_g[2][urow][j] + bf2f(gp[2 * Hh + c0 + j]);
      float opre = lds_g[3][urow][j] + bf2f(gp[3 * Hh + c0 + j]);
      float ig = sigmf(ipre);
      float fg = sigmf(fpre);
      float gg = tanhf(gpre);
      float og = sigmf(opre);
      creg[jj] = fg * creg[jj] + ig * gg;
      float h = og * tanhf(creg[jj]);
      __bf16 hb = f2bf(h);
      hnext[(size_t)urow * Hh + c0 + j] = hb;
      hs[c0 + j] = hb;
      if (t == Tt - 1) {
        hT[(size_t)urow * Hh + c0 + j] = h;
        cT[(size_t)urow * Hh + c0 + j] = creg[jj];
      }
    }

    // ---- inter-WG barrier (release stores, acquire before next reads) -----
    __threadfence();
    __syncthreads();
    if (tid == 0) {
      __hip_atomic_fetch_add(counter, 1u, __ATOMIC_RELEASE,
                             __HIP_MEMORY_SCOPE_AGENT);
      unsigned target = (unsigned)NWG_LSTM * (unsigned)(t + 1);
      while (__hip_atomic_load(counter, __ATOMIC_ACQUIRE,
                               __HIP_MEMORY_SCOPE_AGENT) < target) {
        __builtin_amdgcn_s_sleep(2);
      }
    }
    __syncthreads();
    __builtin_amdgcn_fence(__ATOMIC_ACQUIRE, "agent");
  }
}

// ---------------- LayerNorm over H=256 (one wave per row) -------------------
__global__ void layernorm_kernel(const __bf16* __restrict__ Hs,
                                 const float* __restrict__ g,
                                 const float* __restrict__ b,
                                 __bf16* __restrict__ out) {
  int w = threadIdx.x >> 5, lane = threadIdx.x & 31;
  int row = blockIdx.x * 8 + w;
  const __bf16* p = Hs + (size_t)row * Hh + lane * 8;
  float v[8];
  float s = 0.f;
  #pragma unroll
  for (int j = 0; j < 8; ++j) { v[j] = bf2f(p[j]); s += v[j]; }
  #pragma unroll
  for (int off = 16; off; off >>= 1) s += __shfl_xor(s, off, 32);
  float mu = s * (1.0f / Hh);
  float q = 0.f;
  #pragma unroll
  for (int j = 0; j < 8; ++j) { float d = v[j] - mu; q += d * d; }
  #pragma unroll
  for (int off = 16; off; off >>= 1) q += __shfl_xor(q, off, 32);
  float rstd = rsqrtf(q * (1.0f / Hh) + 1e-5f);
  __bf16* o = out + (size_t)row * Hh + lane * 8;
  #pragma unroll
  for (int j = 0; j < 8; ++j) {
    int col = lane * 8 + j;
    o[j] = f2bf((v[j] - mu) * rstd * g[col] + b[col]);
  }
}

// ---------------------------------------------------------------------------
extern "C" void kernel_launch(void* const* d_in, const int* in_sizes, int n_in,
                              void* d_out, int out_size, void* d_ws,
                              size_t ws_size, hipStream_t stream) {
  (void)in_sizes; (void)n_in; (void)out_size; (void)ws_size;

  const int*   x      = (const int*)  d_in[0];
  const float* embed  = (const float*)d_in[1];
  const float* W_ih0  = (const float*)d_in[2];
  const float* W_hh0  = (const float*)d_in[3];
  const float* b_ih0  = (const float*)d_in[4];
  const float* b_hh0  = (const float*)d_in[5];
  const float* W_ih1  = (const float*)d_in[6];
  const float* W_hh1  = (const float*)d_in[7];
  const float* b_ih1  = (const float*)d_in[8];
  const float* b_hh1  = (const float*)d_in[9];
  const float* ln_g   = (const float*)d_in[10];
  const float* ln_b   = (const float*)d_in[11];
  const float* W_out  = (const float*)d_in[12];
  const float* b_out  = (const float*)d_in[13];

  // ---- workspace layout (256B aligned) ----
  char* ws = (char*)d_ws;
  size_t off = 0;
  auto alloc = [&](size_t bytes) {
    void* p = ws + off;
    off += (bytes + 255) & ~(size_t)255;
    return p;
  };
  __bf16* Wt_ih0 = (__bf16*)alloc((size_t)Ee * G4 * 2);     // [128][1024]
  __bf16* Wt_hh0 = (__bf16*)alloc((size_t)Hh * G4 * 2);     // [256][1024]
  __bf16* Wt_ih1 = (__bf16*)alloc((size_t)Hh * G4 * 2);
  __bf16* Wt_hh1 = (__bf16*)alloc((size_t)Hh * G4 * 2);
  __bf16* Wt_out = (__bf16*)alloc((size_t)Hh * Vv * 2);     // [256][2048]
  __bf16* xemb   = (__bf16*)alloc((size_t)Mrows * Ee * 2);  // 16 MB
  __bf16* Gbuf   = (__bf16*)alloc((size_t)Mrows * G4 * 2);  // 128 MB (g0/g1)
  __bf16* Hs0    = (__bf16*)alloc((size_t)Mrows * Hh * 2);  // 32 MB
  __bf16* Hs1    = (__bf16*)alloc((size_t)Mrows * Hh * 2);  // 32 MB
  __bf16* hcur   = (__bf16*)alloc((size_t)2 * Bb * Hh * 2); // 64 KB
  unsigned* cnts = (unsigned*)alloc(256);

  // ---- output layout: logits [B,T,V] fp32, then h_stack[2,B,H], c_stack ----
  float* out_logits = (float*)d_out;
  float* h_stack = out_logits + (size_t)Mrows * Vv;
  float* c_stack = h_stack + (size_t)2 * Bb * Hh;

  // 1) init barrier counters + h state
  zero_state_kernel<<<128, 256, 0, stream>>>(hcur, cnts, 2);

  // 2) weight convert + transpose to K-major bf16
  transpose_bf16_kernel<<<(G4 * Ee + 255) / 256, 256, 0, stream>>>(W_ih0, Wt_ih0, G4, Ee);
  transpose_bf16_kernel<<<(G4 * Hh + 255) / 256, 256, 0, stream>>>(W_hh0, Wt_hh0, G4, Hh);
  transpose_bf16_kernel<<<(G4 * Hh + 255) / 256, 256, 0, stream>>>(W_ih1, Wt_ih1, G4, Hh);
  transpose_bf16_kernel<<<(G4 * Hh + 255) / 256, 256, 0, stream>>>(W_hh1, Wt_hh1, G4, Hh);
  transpose_bf16_kernel<<<(Vv * Hh + 255) / 256, 256, 0, stream>>>(W_out, Wt_out, Vv, Hh);

  // 3) embedding gather -> bf16
  embed_kernel<<<(Mrows * 32 + 255) / 256, 256, 0, stream>>>(x, embed, xemb);

  // 4) g0 = xemb @ W_ih0^T + b_ih0 + b_hh0   (bf16 out)
  gemm_wmma_kernel<true><<<dim3(G4 / 128, Mrows / 64), 256, 0, stream>>>(
      xemb, Wt_ih0, b_ih0, b_hh0, Gbuf, Mrows, G4, Ee);

  // 5) LSTM layer 0 (persistent, 16 WGs)
  lstm_layer_kernel<<<NWG_LSTM, 256, 0, stream>>>(
      Gbuf, Wt_hh0, Hs0, hcur, cnts + 0, h_stack + 0, c_stack + 0);

  // 6) re-zero h state for layer 1
  zero_state_kernel<<<128, 256, 0, stream>>>(hcur, cnts, 0);

  // 7) g1 = Hs0 @ W_ih1^T + b_ih1 + b_hh1   (bf16 out, reuse Gbuf)
  gemm_wmma_kernel<true><<<dim3(G4 / 128, Mrows / 64), 256, 0, stream>>>(
      Hs0, Wt_ih1, b_ih1, b_hh1, Gbuf, Mrows, G4, Hh);

  // 8) LSTM layer 1
  lstm_layer_kernel<<<NWG_LSTM, 256, 0, stream>>>(
      Gbuf, Wt_hh1, Hs1, hcur, cnts + 1,
      h_stack + (size_t)Bb * Hh, c_stack + (size_t)Bb * Hh);

  // 9) LayerNorm over H (output reuses Hs0)
  layernorm_kernel<<<Mrows / 8, 256, 0, stream>>>(Hs1, ln_g, ln_b, Hs0);

  // 10) logits = normed @ W_out^T + b_out  (fp32 nontemporal stores to d_out)
  gemm_wmma_kernel<false><<<dim3(Vv / 128, Mrows / 64), 256, 0, stream>>>(
      Hs0, Wt_out, b_out, nullptr, out_logits, Mrows, Vv, Hh);
}
